// MultiheadSelfAttention_27023934227121
// MI455X (gfx1250) — matrix operand order repaired and checked by reading, not compiled, not run on hardware
//
#include <hip/hip_runtime.h>

// ---------------------------------------------------------------------------
// Types for gfx1250 WMMA (wave32): bf16 A/B fragments (v16bf), f32 C/D (v8f)
// ---------------------------------------------------------------------------
typedef __bf16 bf16_t;
typedef bf16_t v8bf  __attribute__((ext_vector_type(8)));
typedef bf16_t v16bf __attribute__((ext_vector_type(16)));
typedef float  v8f   __attribute__((ext_vector_type(8)));

#define DEV static __device__ __forceinline__

// CDNA5 async global->LDS copy (ASYNCcnt path), if the toolchain exposes it.
#if defined(__has_builtin)
#  if __has_builtin(__builtin_amdgcn_global_load_async_to_lds_b128)
#    define USE_ASYNC_LDS 1
#  endif
#endif
#ifndef USE_ASYNC_LDS
#  define USE_ASYNC_LDS 0
#endif

#if USE_ASYNC_LDS
// Exact pointee type the builtin expects: GCC-style v4i (per hipcc diagnostic)
typedef int v4i_gcc __attribute__((vector_size(16)));
typedef __attribute__((address_space(1))) v4i_gcc g_v4i;   // global
typedef __attribute__((address_space(3))) v4i_gcc l_v4i;   // LDS
#endif

// Problem constants
#define BB 4
#define TT 2048
#define CC 1024
#define HH 16
#define DD 64
#define MM (BB * TT)   // 8192

// Workspace layout (bytes)
static constexpr size_t SZ_XB  = (size_t)MM * CC * 2;        // x (and later y) in bf16
static constexpr size_t SZ_W   = (size_t)CC * CC * 2;        // one transposed bf16 weight
static constexpr size_t OFF_W  = SZ_XB;
static constexpr size_t SZ_QKV = (size_t)BB * HH * TT * DD * 2;
static constexpr size_t OFF_Q  = OFF_W + 4 * SZ_W;
static constexpr size_t OFF_K  = OFF_Q + SZ_QKV;
static constexpr size_t OFF_V  = OFF_K + SZ_QKV;

DEV v8f wmma_bf16(v16bf a, v16bf b, v8f c) {
  return __builtin_amdgcn_wmma_f32_16x16x32_bf16(
      /*neg_a=*/false, a, /*neg_b=*/false, b,
      /*c_mod=*/(short)0, c, /*reuse_a=*/false, /*reuse_b=*/false);
}

// A-fragment (16x32, MxK, row-major source, leading dim `ld` elements).
// ISA layout: lanes 0-15 row=l, K={0..7,16..23}; lanes 16-31 row=l, K={8..15,24..31}
DEV v16bf load_a_frag(const bf16_t* base, int ld) {
  const int lane = threadIdx.x & 31;
  const int l = lane & 15, h = lane >> 4;
  const bf16_t* p = base + (size_t)l * ld + h * 8;
  v8bf lo = *(const v8bf*)p;
  v8bf hi = *(const v8bf*)(p + 16);
  v16bf a;
#pragma unroll
  for (int i = 0; i < 8; ++i) { a[i] = lo[i]; a[i + 8] = hi[i]; }
  return a;
}

// B-fragment (32x16, KxN) loaded from B^T stored row-major [N, K] with ld `ld`.
// Layout: lanes 0-15 col=l, K={0..7,16..23}; lanes 16-31 col=l, K={8..15,24..31}
DEV v16bf load_b_frag(const bf16_t* base, int ld) {
  const int lane = threadIdx.x & 31;
  const int l = lane & 15, h = lane >> 4;
  const bf16_t* p = base + (size_t)l * ld + h * 8;
  v8bf lo = *(const v8bf*)p;
  v8bf hi = *(const v8bf*)(p + 16);
  v16bf b;
#pragma unroll
  for (int i = 0; i < 8; ++i) { b[i] = lo[i]; b[i + 8] = hi[i]; }
  return b;
}

// Butterfly reductions within each 16-lane half (rows of a C-tile)
DEV float half_max(float v) {
#pragma unroll
  for (int m = 8; m > 0; m >>= 1) v = fmaxf(v, __shfl_xor(v, m, 32));
  return v;
}
DEV float half_sum(float v) {
#pragma unroll
  for (int m = 8; m > 0; m >>= 1) v += __shfl_xor(v, m, 32);
  return v;
}

// ---------------------------------------------------------------------------
// Elementwise f32 -> bf16
// ---------------------------------------------------------------------------
__global__ void __launch_bounds__(256) cvt_kernel(const float* __restrict__ in,
                                                  bf16_t* __restrict__ out, int n) {
  int i = blockIdx.x * 256 + threadIdx.x;
  if (i < n) out[i] = (bf16_t)in[i];
}

// W [K=1024, N=1024] f32 -> W^T [N, K] bf16 (coalesced writes)
__global__ void __launch_bounds__(256) transpose_kernel(const float* __restrict__ in,
                                                        bf16_t* __restrict__ out) {
  int i = blockIdx.x * 256 + threadIdx.x;   // over 1M outputs
  int n = i >> 10, k = i & 1023;
  out[i] = (bf16_t)in[k * CC + n];
}

// ---------------------------------------------------------------------------
// GEMM: out = A[M=8192,K=1024] @ B[K=1024,N=1024] + bias, via bf16 WMMA.
// Bt is B^T (row-major [N,K], bf16). Each wave computes a 32x64 tile:
// 8 v_wmma_f32_16x16x32_bf16 per 32-deep K step, 32 K steps.
// mode 0/1: write bf16 [B,H,T,D] (Q scaled by 1/8 / K)
// mode 2  : write bf16 [B,H,D,T] (V transposed for the P@V B-fragment)
// mode 3  : write f32  [M, N]    (final output projection)
// ---------------------------------------------------------------------------
__global__ void __launch_bounds__(256) gemm_kernel(const bf16_t* __restrict__ A,
                                                   const bf16_t* __restrict__ Bt,
                                                   const float* __restrict__ bias,
                                                   void* __restrict__ outp,
                                                   int mode, float scale) {
  const int wid = blockIdx.x * 8 + (threadIdx.x >> 5);
  const int m0 = (wid >> 4) * 32;           // 256 row-tiles of 32
  const int n0 = (wid & 15) * 64;           // 16 col-tiles of 64
  const int lane = threadIdx.x & 31;
  const int l = lane & 15, half = lane >> 4;

  v8f c[2][4];
#pragma unroll
  for (int p = 0; p < 2; ++p)
#pragma unroll
    for (int j = 0; j < 4; ++j)
#pragma unroll
      for (int r = 0; r < 8; ++r) c[p][j][r] = 0.f;

  const bf16_t* Arow0 = A + (size_t)m0 * CC;
  const bf16_t* Arow1 = A + (size_t)(m0 + 16) * CC;

  for (int kb = 0; kb < CC; kb += 32) {
    v16bf a0 = load_a_frag(Arow0 + kb, CC);
    v16bf a1 = load_a_frag(Arow1 + kb, CC);
#pragma unroll
    for (int j = 0; j < 4; ++j) {
      v16bf bf = load_b_frag(Bt + (size_t)(n0 + j * 16) * CC + kb, CC);
      c[0][j] = wmma_bf16(a0, bf, c[0][j]);
      c[1][j] = wmma_bf16(a1, bf, c[1][j]);
    }
  }

#pragma unroll
  for (int j = 0; j < 4; ++j) {
    const int gcol = n0 + j * 16 + l;
    const float bv = bias[gcol];
#pragma unroll
    for (int p = 0; p < 2; ++p) {
#pragma unroll
      for (int r = 0; r < 8; ++r) {
        const int grow = m0 + p * 16 + r + 8 * half;     // global token index
        const float val = (c[p][j][r] + bv) * scale;
        if (mode == 3) {
          ((float*)outp)[(size_t)grow * CC + gcol] = val;
        } else {
          const int bb = grow >> 11, tt = grow & (TT - 1);
          const int h2 = gcol >> 6, dd = gcol & (DD - 1);
          bf16_t* ob = (bf16_t*)outp;
          if (mode == 2)   // V transposed: [B,H,D,T]
            ob[(((size_t)(bb * HH + h2)) * DD + dd) * TT + tt] = (bf16_t)val;
          else             // Q/K: [B,H,T,D]
            ob[(((size_t)(bb * HH + h2)) * TT + tt) * DD + dd] = (bf16_t)val;
        }
      }
    }
  }
}

// ---------------------------------------------------------------------------
// Stage one 32-key step of K (32x64, contiguous) and V^T window (64 rows x 32)
// into LDS. 256 threads, one b128 transfer each per tile. Uses the CDNA5
// async global->LDS path (ASYNCcnt) when available.
// ---------------------------------------------------------------------------
DEV void stage_kv(bf16_t* kdst, bf16_t* vdst,
                  const bf16_t* Kp, const bf16_t* Vp, int s0) {
  const int t = threadIdx.x;                       // 0..255
  const bf16_t* gk = Kp + (size_t)s0 * DD + t * 8; // K tile: 4KB contiguous
  bf16_t*       lk = kdst + t * 8;
  const int row = t >> 2, part = t & 3;            // V^T: 64 rows x 64B
  const bf16_t* gv = Vp + (size_t)row * TT + s0 + part * 8;
  bf16_t*       lv = vdst + row * 32 + part * 8;
#if USE_ASYNC_LDS
  __builtin_amdgcn_global_load_async_to_lds_b128(
      (g_v4i*)(uintptr_t)gk, (l_v4i*)(uintptr_t)lk, 0, 0);
  __builtin_amdgcn_global_load_async_to_lds_b128(
      (g_v4i*)(uintptr_t)gv, (l_v4i*)(uintptr_t)lv, 0, 0);
#else
  *(v8bf*)lk = *(const v8bf*)gk;
  *(v8bf*)lv = *(const v8bf*)gv;
#endif
}

DEV void stage_wait_and_sync() {
#if USE_ASYNC_LDS
#  if defined(__has_builtin) && __has_builtin(__builtin_amdgcn_s_wait_asynccnt)
  __builtin_amdgcn_s_wait_asynccnt(0);
#  else
  asm volatile("s_wait_asynccnt 0" ::: "memory");
#  endif
#endif
  __syncthreads();
}

// ---------------------------------------------------------------------------
// Flash attention: block = 8 waves sharing one (b,h); wave = 16-query tile.
// K/V for each 32-key step are staged once per block into LDS (double
// buffered), giving 8x reuse; fragments then come from LDS (ds_load_b128).
// Q pre-scaled by 1/sqrt(D). Output written bf16 into y[B,T,C].
// ---------------------------------------------------------------------------
__global__ void __launch_bounds__(256) attn_kernel(const bf16_t* __restrict__ Q,
                                                   const bf16_t* __restrict__ K,
                                                   const bf16_t* __restrict__ Vt,
                                                   bf16_t* __restrict__ Y) {
  __shared__ __align__(16) bf16_t kbuf[2][32 * DD];     // 2 x 4KB
  __shared__ __align__(16) bf16_t vbuf[2][DD * 32];     // 2 x 4KB
  __shared__ __align__(16) bf16_t pstage[8][16 * 32];   // per-wave P staging, 8KB

  const int warp = threadIdx.x >> 5;
  const int wid  = blockIdx.x * 8 + warp;               // 0..8191
  const int bh   = wid >> 7;                            // same for all 8 waves in block
  const int qt   = wid & 127;                           // query tile
  const int b    = bh >> 4, hh = bh & 15;
  const int lane = threadIdx.x & 31;
  const int l = lane & 15, half = lane >> 4;

  const bf16_t* Qp = Q  + ((size_t)bh * TT + qt * 16) * DD;
  const bf16_t* Kp = K  + (size_t)bh * TT * DD;
  const bf16_t* Vp = Vt + (size_t)bh * DD * TT;
  bf16_t*       Yp = Y  + ((size_t)(b * TT + qt * 16)) * CC + hh * DD;

  const v16bf qa0 = load_a_frag(Qp, DD);        // K-dim d=0..31
  const v16bf qa1 = load_a_frag(Qp + 32, DD);   // K-dim d=32..63

  v8f acc[4];
  float m[8], lsum[8];
#pragma unroll
  for (int j = 0; j < 4; ++j)
#pragma unroll
    for (int r = 0; r < 8; ++r) acc[j][r] = 0.f;
#pragma unroll
  for (int r = 0; r < 8; ++r) { m[r] = -3.0e38f; lsum[r] = 0.f; }

  bf16_t* pst = &pstage[warp][0];

  stage_kv(kbuf[0], vbuf[0], Kp, Vp, 0);
  stage_wait_and_sync();

  int buf = 0;
  for (int s0 = 0; s0 < TT; s0 += 32) {
    if (s0 + 32 < TT)                                  // prefetch next step
      stage_kv(kbuf[buf ^ 1], vbuf[buf ^ 1], Kp, Vp, s0 + 32);

    const bf16_t* kb = kbuf[buf];
    const bf16_t* vb = vbuf[buf];

    v8f sc0 = {0,0,0,0,0,0,0,0};
    v8f sc1 = {0,0,0,0,0,0,0,0};
    sc0 = wmma_bf16(qa0, load_b_frag(kb, DD), sc0);
    sc0 = wmma_bf16(qa1, load_b_frag(kb + 32, DD), sc0);
    sc1 = wmma_bf16(qa0, load_b_frag(kb + 16 * DD, DD), sc1);
    sc1 = wmma_bf16(qa1, load_b_frag(kb + 16 * DD + 32, DD), sc1);

    // online softmax over the 32 new keys
#pragma unroll
    for (int r = 0; r < 8; ++r) {
      float v = half_max(fmaxf(sc0[r], sc1[r]));
      float mnew = fmaxf(m[r], v);
      float corr = __expf(m[r] - mnew);
      m[r] = mnew;
      float p0 = __expf(sc0[r] - mnew);
      float p1 = __expf(sc1[r] - mnew);
      pst[(r + 8 * half) * 32 + l]      = (bf16_t)p0;   // row-major 16x32 P tile
      pst[(r + 8 * half) * 32 + 16 + l] = (bf16_t)p1;
      lsum[r] = lsum[r] * corr + half_sum(p0 + p1);
#pragma unroll
      for (int j = 0; j < 4; ++j) acc[j][r] *= corr;
    }
    asm volatile("s_wait_dscnt 0" ::: "memory");        // P stores visible in-wave
    v16bf pa = load_a_frag(pst, 32);                    // C-layout -> A-fragment
#pragma unroll
    for (int j = 0; j < 4; ++j)
      acc[j] = wmma_bf16(pa, load_b_frag(vb + (size_t)(j * 16) * 32, 32), acc[j]);

    stage_wait_and_sync();   // next buffer landed; all reads of `buf` complete
    buf ^= 1;
  }

#pragma unroll
  for (int r = 0; r < 8; ++r) {
    const float inv = 1.0f / lsum[r];
#pragma unroll
    for (int j = 0; j < 4; ++j)
      Yp[(size_t)(r + 8 * half) * CC + j * 16 + l] = (bf16_t)(acc[j][r] * inv);
  }
}

// ---------------------------------------------------------------------------
extern "C" void kernel_launch(void* const* d_in, const int* in_sizes, int n_in,
                              void* d_out, int out_size, void* d_ws, size_t ws_size,
                              hipStream_t stream) {
  const float* x  = (const float*)d_in[0];
  const float* Wq = (const float*)d_in[1];
  const float* bq = (const float*)d_in[2];
  const float* Wk = (const float*)d_in[3];
  const float* bk = (const float*)d_in[4];
  const float* Wv = (const float*)d_in[5];
  const float* bv = (const float*)d_in[6];
  const float* Wo = (const float*)d_in[7];
  const float* bo = (const float*)d_in[8];

  char* ws = (char*)d_ws;
  bf16_t* xb  = (bf16_t*)ws;                      // x bf16; reused as y after attention
  bf16_t* wqT = (bf16_t*)(ws + OFF_W + 0 * SZ_W);
  bf16_t* wkT = (bf16_t*)(ws + OFF_W + 1 * SZ_W);
  bf16_t* wvT = (bf16_t*)(ws + OFF_W + 2 * SZ_W);
  bf16_t* woT = (bf16_t*)(ws + OFF_W + 3 * SZ_W);
  bf16_t* Qb  = (bf16_t*)(ws + OFF_Q);
  bf16_t* Kb  = (bf16_t*)(ws + OFF_K);
  bf16_t* Vb  = (bf16_t*)(ws + OFF_V);

  const int NX = MM * CC;                         // 8388608
  cvt_kernel<<<NX / 256, 256, 0, stream>>>(x, xb, NX);
  transpose_kernel<<<(CC * CC) / 256, 256, 0, stream>>>(Wq, wqT);
  transpose_kernel<<<(CC * CC) / 256, 256, 0, stream>>>(Wk, wkT);
  transpose_kernel<<<(CC * CC) / 256, 256, 0, stream>>>(Wv, wvT);
  transpose_kernel<<<(CC * CC) / 256, 256, 0, stream>>>(Wo, woT);

  // waves = (8192/32)*(1024/64) = 4096 -> 512 blocks of 8 waves
  gemm_kernel<<<512, 256, 0, stream>>>(xb, wqT, bq, Qb, 0, 0.125f); // Q * 1/sqrt(64)
  gemm_kernel<<<512, 256, 0, stream>>>(xb, wkT, bk, Kb, 1, 1.0f);
  gemm_kernel<<<512, 256, 0, stream>>>(xb, wvT, bv, Vb, 2, 1.0f);

  // waves = B*H*(T/16) = 8192 -> 1024 blocks of 8 waves
  attn_kernel<<<1024, 256, 0, stream>>>(Qb, Kb, Vb, xb);

  gemm_kernel<<<512, 256, 0, stream>>>(xb, woT, bo, d_out, 3, 1.0f);
}
